// HydraLayer_14431090115014
// MI455X (gfx1250) — matrix operand order repaired and checked by reading, not compiled, not run on hardware
//
#include <hip/hip_runtime.h>
#include <hip/hip_bf16.h>

// ---------------- problem constants ----------------
#define B_       4
#define L_       2048
#define DM_      1024
#define DIN_     2048          // EXPAND*DM
#define NH_      32
#define HEADDIM_ 64
#define DSTATE_  64
#define DCONV_   7
#define CONVDIM_ 2176          // DIN + 2*DSTATE
#define DPROJ_   4288          // 2*DIN + 2*DSTATE + 2*NH
#define MTOT_    (B_ * L_)     // 8192 rows
#define EPS_     1e-5f

// ---------------- WMMA types ----------------
typedef __bf16 v16bf __attribute__((ext_vector_type(16)));
typedef float  v8f   __attribute__((ext_vector_type(8)));

union FragU { v16bf v; unsigned int u[8]; };

__device__ __forceinline__ unsigned short f2bf(float f) {
  unsigned int u = __float_as_uint(f);
  unsigned int r = u + 0x7FFFu + ((u >> 16) & 1u);
  return (unsigned short)(r >> 16);
}

__device__ __forceinline__ float silu(float v) { return v / (1.f + __expf(-v)); }

// A-matrix 16x32 bf16 fragment at K-chunk kc: interleaved K layout (ISA 7.12.2).
__device__ __forceinline__ v16bf load_frag_a(const unsigned short* smem,
                                             int row0, int kc, int ldk, int lane) {
  int r    = row0 + (lane & 15);
  int half = (lane >> 4) & 1;
  const unsigned int* rp = (const unsigned int*)(smem + (size_t)r * ldk) + (kc >> 1);
  FragU f;
#pragma unroll
  for (int g = 0; g < 8; ++g) {
    int kk = (g < 4) ? (half * 8 + 2 * g) : (16 + half * 8 + 2 * (g - 4));
    f.u[g] = rp[kk >> 1];
  }
  return f.v;
}

// B-matrix 32x16 bf16 fragment: lane = column, interleaved K layout matching A.
__device__ __forceinline__ v16bf load_frag_b(const unsigned short* smem,
                                             int col0, int kc, int ldk, int lane) {
  int n    = col0 + (lane & 15);
  int half = (lane >> 4) & 1;
  const unsigned int* rp = (const unsigned int*)(smem + (size_t)n * ldk) + (kc >> 1);
  FragU f;
#pragma unroll
  for (int g = 0; g < 8; ++g) f.u[g] = rp[(g < 4) ? (half * 4 + g) : (8 + half * 4 + (g - 4))];
  return f.v;
}

#define WMMA_BF16(a, b, c) \
  __builtin_amdgcn_wmma_f32_16x16x32_bf16(false, (a), false, (b), (short)0, (c), false, false)

// ---------------- bf16 WMMA GEMM:  D = A(MxK) * B(NxK)^T (+ Cadd if HASC) -------
// Block tile 128x64, BK=64, double-buffered LDS, global->reg prefetch of next
// tile issued before the WMMAs of the current tile (one barrier per K-tile).
#define BM 128
#define BN 64
#define BK 64
#define LDK 68   // padded LDS row (bf16 elems): 136B rows, 8B aligned uint2 stores

template <bool HASC>
__global__ __launch_bounds__(256) void gemm_bf16_wmma(
    const unsigned short* __restrict__ A,   // [M,K] bf16 row-major
    const unsigned short* __restrict__ Bw,  // [N,K] bf16 row-major
    const float* __restrict__ Cadd,         // [M,N] f32 (used iff HASC)
    float* __restrict__ Dout,               // [M,N] f32
    int M, int N, int K)
{
  __shared__ unsigned short As[2][BM][LDK];
  __shared__ unsigned short Bs[2][BN][LDK];

  const int tid  = threadIdx.x;
  const int lane = tid & 31;
  const int wave = tid >> 5;     // 0..7
  const int wm   = wave >> 1;    // 0..3  -> 32-row slab
  const int wn   = wave & 1;     // 0..1  -> 32-col slab
  const int m0   = blockIdx.y * BM;
  const int n0   = blockIdx.x * BN;

  uint2 areg[8], breg[4];        // staged global data for one K-tile

  auto load_regs = [&](int k0) {
#pragma unroll
    for (int j = 0; j < 8; ++j) {            // A: 128 rows x 64 bf16 = 2048 uint2
      int i = tid + 256 * j;
      int r = i >> 4, c = i & 15;            // 16 uint2 per row
      areg[j] = *(const uint2*)&A[(size_t)(m0 + r) * K + k0 + c * 4];
    }
#pragma unroll
    for (int j = 0; j < 4; ++j) {            // B: 64 rows x 64 bf16 = 1024 uint2
      int i = tid + 256 * j;
      int r = i >> 4, c = i & 15;
      breg[j] = *(const uint2*)&Bw[(size_t)(n0 + r) * K + k0 + c * 4];
    }
  };
  auto store_lds = [&](int buf) {
#pragma unroll
    for (int j = 0; j < 8; ++j) {
      int i = tid + 256 * j;
      int r = i >> 4, c = i & 15;
      *(uint2*)&As[buf][r][c * 4] = areg[j];
    }
#pragma unroll
    for (int j = 0; j < 4; ++j) {
      int i = tid + 256 * j;
      int r = i >> 4, c = i & 15;
      *(uint2*)&Bs[buf][r][c * 4] = breg[j];
    }
  };

  v8f acc[2][2] = {};
  const int nk = K / BK;

  load_regs(0);
  store_lds(0);
  __syncthreads();

  for (int kt = 0; kt < nk; ++kt) {
    if (kt + 1 < nk) load_regs((kt + 1) * BK);          // in-flight during WMMAs
    if (kt + 2 < nk) {                                   // L2 prefetch of tile k+2
      __builtin_prefetch(&A[(size_t)(m0 + (tid >> 1)) * K + (size_t)(kt + 2) * BK], 0, 3);
      __builtin_prefetch(&Bw[(size_t)(n0 + (tid & 63)) * K + (size_t)(kt + 2) * BK], 0, 3);
    }
    const int cur = kt & 1;
#pragma unroll
    for (int kc = 0; kc < BK; kc += 32) {
      v16bf a0 = load_frag_a(&As[cur][0][0], wm * 32,      kc, LDK, lane);
      v16bf a1 = load_frag_a(&As[cur][0][0], wm * 32 + 16, kc, LDK, lane);
      v16bf b0 = load_frag_b(&Bs[cur][0][0], wn * 32,      kc, LDK, lane);
      v16bf b1 = load_frag_b(&Bs[cur][0][0], wn * 32 + 16, kc, LDK, lane);
      acc[0][0] = WMMA_BF16(a0, b0, acc[0][0]);
      acc[0][1] = WMMA_BF16(a0, b1, acc[0][1]);
      acc[1][0] = WMMA_BF16(a1, b0, acc[1][0]);
      acc[1][1] = WMMA_BF16(a1, b1, acc[1][1]);
    }
    if (kt + 1 < nk) store_lds(1 - cur);                // disjoint buffer from 'cur'
    __syncthreads();
  }

  // C/D layout: VGPR v, lanes 0-15 -> M=v, N=lane ; lanes 16-31 -> M=v+8, N=lane-16
  const int hi = (lane >> 4) & 1;
  const int nn = lane & 15;
#pragma unroll
  for (int im = 0; im < 2; ++im)
#pragma unroll
    for (int jn = 0; jn < 2; ++jn) {
      size_t base = (size_t)(m0 + wm * 32 + im * 16 + hi * 8) * N
                  + (size_t)(n0 + wn * 32 + jn * 16 + nn);
      float* dp = Dout + base;
      const float* cp = Cadd + base;
#pragma unroll
      for (int v = 0; v < 8; ++v) {
        float r = acc[im][jn][v];
        if (HASC) r += cp[(size_t)v * N];
        dp[(size_t)v * N] = r;
      }
    }
}

// ---------------- fp32 -> bf16 convert ----------------
__global__ void cvt_bf16_kernel(const float* __restrict__ src,
                                unsigned short* __restrict__ dst, int n) {
  int i = blockIdx.x * 256 + threadIdx.x;
  if (i < n) dst[i] = f2bf(src[i]);
}

// ---------------- LayerNorm over DM, output bf16 ----------------
__global__ __launch_bounds__(256) void ln_kernel(
    const float* __restrict__ x, const float* __restrict__ g,
    const float* __restrict__ be, unsigned short* __restrict__ ubf) {
  int row = blockIdx.x, t = threadIdx.x;
  __shared__ float red[256], red2[256];
  float v[4], s = 0.f, sq = 0.f;
#pragma unroll
  for (int k = 0; k < 4; ++k) {
    v[k] = x[(size_t)row * DM_ + t + 256 * k];
    s += v[k]; sq += v[k] * v[k];
  }
  red[t] = s; red2[t] = sq; __syncthreads();
  for (int st = 128; st > 0; st >>= 1) {
    if (t < st) { red[t] += red[t + st]; red2[t] += red2[t + st]; }
    __syncthreads();
  }
  float mu  = red[0] * (1.f / DM_);
  float var = red2[0] * (1.f / DM_) - mu * mu;
  float inv = rsqrtf(var + EPS_);
#pragma unroll
  for (int k = 0; k < 4; ++k) {
    int c = t + 256 * k;
    ubf[(size_t)row * DM_ + c] = f2bf((v[k] - mu) * inv * g[c] + be[c]);
  }
}

// ---------------- depthwise conv7 (same pad) + bias + SiLU, split xs/B/C -------
__global__ void conv_silu_kernel(const float* __restrict__ zx,
                                 const float* __restrict__ cw,
                                 const float* __restrict__ cb,
                                 float* __restrict__ xs,
                                 float* __restrict__ Bm,
                                 float* __restrict__ Cm) {
  size_t idx = (size_t)blockIdx.x * 256 + threadIdx.x;
  if (idx >= (size_t)MTOT_ * CONVDIM_) return;
  int c   = (int)(idx % CONVDIM_);
  int row = (int)(idx / CONVDIM_);
  int b = row / L_, l = row % L_;
  float acc = cb[c];
#pragma unroll
  for (int j = 0; j < DCONV_; ++j) {
    int ll = l + j - 3;
    if (ll >= 0 && ll < L_)
      acc += zx[((size_t)b * L_ + ll) * DPROJ_ + DIN_ + c] * cw[c * DCONV_ + j];
  }
  float v = silu(acc);
  if (c < DIN_)                xs[(size_t)row * DIN_ + c] = v;
  else if (c < DIN_ + DSTATE_) Bm[(size_t)row * DSTATE_ + (c - DIN_)] = v;
  else                         Cm[(size_t)row * DSTATE_ + (c - DIN_ - DSTATE_)] = v;
}

// ---------------- dt2 = softplus(dt(+flip) + bias), dA = exp(dt2 * -exp(A_log)) --
__global__ void dtda_kernel(const float* __restrict__ zx,
                            const float* __restrict__ dtb,
                            const float* __restrict__ Alog,
                            float* __restrict__ dt2a, float* __restrict__ dAa) {
  int idx = blockIdx.x * 256 + threadIdx.x;   // [2B][L][NH]
  int h = idx & (NH_ - 1);
  int l = (idx >> 5) & (L_ - 1);
  int d = idx >> 16;                           // / (L*NH) = /65536
  float raw;
  if (d < B_)
    raw = zx[((size_t)d * L_ + l) * DPROJ_ + (DIN_ + CONVDIM_) + h];
  else
    raw = zx[((size_t)(d - B_) * L_ + (L_ - 1 - l)) * DPROJ_ + (DIN_ + CONVDIM_) + NH_ + h];
  float xv = raw + dtb[h];
  float dtv = (xv > 20.f) ? xv : log1pf(__expf(xv));
  float dA = __expf(dtv * (-__expf(Alog[h])));
  dt2a[idx] = dtv;
  dAa[idx]  = dA;
}

// ---------------- sequential SSM scan: one block per (direction*batch, head) ----
__global__ __launch_bounds__(256) void scan_kernel(
    const float* __restrict__ xs, const float* __restrict__ Bm,
    const float* __restrict__ Cm, const float* __restrict__ dt2a,
    const float* __restrict__ dAa, float* __restrict__ y2) {
  int blk = blockIdx.x;            // 0 .. 2B*NH-1
  int d = blk >> 5;
  int h = blk & (NH_ - 1);
  int b = (d < B_) ? d : d - B_;
  int t = threadIdx.x;
  int p = t & 63;
  int q = t >> 6;

  __shared__ float xls[64], Bls[64], Cls[64], sc[2];
  __shared__ float ypart[256];

  float hreg[16];
#pragma unroll
  for (int i = 0; i < 16; ++i) hreg[i] = 0.f;

  if (t < 64) y2[((size_t)d * L_) * DIN_ + h * 64 + t] = 0.f;

  for (int l = 0; l < L_; ++l) {
    int lsrc = (d < B_) ? l : (L_ - 1 - l);
    size_t rs = (size_t)b * L_ + lsrc;
    if (t < 64)        xls[t]       = xs[rs * DIN_ + h * 64 + t];
    else if (t < 128)  Bls[t - 64]  = Bm[rs * DSTATE_ + (t - 64)];
    else if (t < 192)  Cls[t - 128] = Cm[rs * DSTATE_ + (t - 128)];
    else if (t == 192) {
      size_t si = ((size_t)d * L_ + l) * NH_ + h;
      sc[0] = dAa[si]; sc[1] = dt2a[si];
    }
    __syncthreads();

    float dA_t = sc[0];
    float xv   = sc[1] * xls[p];
    float acc  = 0.f;
#pragma unroll
    for (int i = 0; i < 16; ++i) {
      int n = q + 4 * i;
      hreg[i] = dA_t * hreg[i] + xv * Bls[n];
      acc += hreg[i] * Cls[n];
    }
    ypart[t] = acc;
    __syncthreads();

    if (t < 64 && (l + 1) < L_) {
      float yv = ypart[t] + ypart[t + 64] + ypart[t + 128] + ypart[t + 192];
      y2[((size_t)d * L_ + (l + 1)) * DIN_ + h * 64 + t] = yv;
    }
  }
}

// ---------------- combine: fwd+bwd, D-path, gate silu(z), RMSNorm -> bf16 -------
__global__ __launch_bounds__(256) void combine_kernel(
    const float* __restrict__ y2, const float* __restrict__ xs,
    const float* __restrict__ zx, const float* __restrict__ fcD,
    const float* __restrict__ Dvec, const float* __restrict__ nw,
    unsigned short* __restrict__ Ybf) {
  int row = blockIdx.x, t = threadIdx.x;
  int b = row / L_, l = row % L_;
  __shared__ float xs_s[DIN_];
  __shared__ float red[256];
  __shared__ float Dd[NH_];

#pragma unroll
  for (int k = 0; k < 8; ++k) xs_s[t + 256 * k] = xs[(size_t)row * DIN_ + t + 256 * k];
  __syncthreads();

  {
    int h = t >> 3, part = t & 7;
    const float* wr = fcD + (size_t)h * DIN_ + part * 256;
    const float* xr = xs_s + part * 256;
    float s = 0.f;
    for (int j = 0; j < 256; ++j) s += xr[j] * wr[j];
    red[t] = s;
  }
  __syncthreads();
  if (t < NH_) {
    float s = 0.f;
#pragma unroll
    for (int k = 0; k < 8; ++k) s += red[t * 8 + k];
    Dd[t] = s + Dvec[t];
  }
  __syncthreads();

  float yv[8], sq = 0.f;
#pragma unroll
  for (int k = 0; k < 8; ++k) {
    int c = t + 256 * k;
    int h = c >> 6;
    float v = y2[((size_t)b * L_ + l) * DIN_ + c]
            + y2[((size_t)(B_ + b) * L_ + (L_ - 1 - l)) * DIN_ + c]
            + xs_s[c] * Dd[h];
    float z = zx[(size_t)row * DPROJ_ + c];
    v *= silu(z);
    yv[k] = v; sq += v * v;
  }
  red[t] = sq; __syncthreads();
  for (int st = 128; st > 0; st >>= 1) {
    if (t < st) red[t] += red[t + st];
    __syncthreads();
  }
  float scale = rsqrtf(red[0] * (1.f / DIN_) + EPS_);
#pragma unroll
  for (int k = 0; k < 8; ++k) {
    int c = t + 256 * k;
    Ybf[(size_t)row * DIN_ + c] = f2bf(yv[k] * scale * nw[c]);
  }
}

// ---------------- launch ----------------
extern "C" void kernel_launch(void* const* d_in, const int* in_sizes, int n_in,
                              void* d_out, int out_size, void* d_ws, size_t ws_size,
                              hipStream_t stream) {
  const float* x      = (const float*)d_in[0];
  const float* ln_g   = (const float*)d_in[1];
  const float* ln_b   = (const float*)d_in[2];
  const float* W_in   = (const float*)d_in[3];
  const float* conv_w = (const float*)d_in[4];
  const float* conv_b = (const float*)d_in[5];
  const float* dt_bias= (const float*)d_in[6];
  const float* A_log  = (const float*)d_in[7];
  const float* fc_D_w = (const float*)d_in[8];
  const float* Dvec   = (const float*)d_in[9];
  const float* norm_w = (const float*)d_in[10];
  const float* W_out  = (const float*)d_in[11];
  float* out = (float*)d_out;

  char* p = (char*)d_ws;
  unsigned short* u_bf   = (unsigned short*)p; p += (size_t)MTOT_ * DM_ * 2;
  unsigned short* Win_bf = (unsigned short*)p; p += (size_t)DPROJ_ * DM_ * 2;
  unsigned short* Wout_bf= (unsigned short*)p; p += (size_t)DM_ * DIN_ * 2;
  float* zx   = (float*)p; p += (size_t)MTOT_ * DPROJ_ * 4;
  float* xs   = (float*)p; p += (size_t)MTOT_ * DIN_ * 4;
  float* Bm   = (float*)p; p += (size_t)MTOT_ * DSTATE_ * 4;
  float* Cm   = (float*)p; p += (size_t)MTOT_ * DSTATE_ * 4;
  float* dt2a = (float*)p; p += (size_t)2 * MTOT_ * NH_ * 4;
  float* dAa  = (float*)p; p += (size_t)2 * MTOT_ * NH_ * 4;
  float* y2   = (float*)p; p += (size_t)2 * MTOT_ * DIN_ * 4;
  unsigned short* Ybf = (unsigned short*)p;

  cvt_bf16_kernel<<<(DPROJ_ * DM_ + 255) / 256, 256, 0, stream>>>(W_in, Win_bf, DPROJ_ * DM_);
  cvt_bf16_kernel<<<(DM_ * DIN_ + 255) / 256, 256, 0, stream>>>(W_out, Wout_bf, DM_ * DIN_);
  ln_kernel<<<MTOT_, 256, 0, stream>>>(x, ln_g, ln_b, u_bf);
  gemm_bf16_wmma<false><<<dim3(DPROJ_ / BN, MTOT_ / BM), 256, 0, stream>>>(
      u_bf, Win_bf, nullptr, zx, MTOT_, DPROJ_, DM_);
  conv_silu_kernel<<<(unsigned)(((size_t)MTOT_ * CONVDIM_ + 255) / 256), 256, 0, stream>>>(
      zx, conv_w, conv_b, xs, Bm, Cm);
  dtda_kernel<<<(2 * MTOT_ * NH_) / 256, 256, 0, stream>>>(zx, dt_bias, A_log, dt2a, dAa);
  scan_kernel<<<2 * B_ * NH_, 256, 0, stream>>>(xs, Bm, Cm, dt2a, dAa, y2);
  combine_kernel<<<MTOT_, 256, 0, stream>>>(y2, xs, zx, fc_D_w, Dvec, norm_w, Ybf);
  gemm_bf16_wmma<true><<<dim3(DM_ / BN, MTOT_ / BM), 256, 0, stream>>>(
      Ybf, Wout_bf, x, out, MTOT_, DM_, DIN_);
}
